// LRP_GraphEmbModule_54374285967908
// MI455X (gfx1250) — matrix operand order repaired and checked by reading, compiled, not run on hardware
//
#include <hip/hip_runtime.h>

typedef __attribute__((ext_vector_type(16))) __bf16 v16bf;
typedef __attribute__((ext_vector_type(8)))  float  v8f;

#define N_NODES 100000
#define NPERM   100000
#define LRP_L   16
#define HID     64
#define NLAYERS 4
#define NGRAPH  1024
#define NTILES  6250      // NPERM/16 == N_NODES/16
#define WPB     8         // waves (wave32) per block
#define TPB     256

// ---------------------------------------------------------------------------
// CDNA5 async global->LDS block copy (GLOBAL_LOAD_ASYNC_TO_LDS_B128, ASYNCcnt).
__device__ __forceinline__ void async_stage(void* lds_dst, const void* gsrc,
                                            int bytes) {
  unsigned int lds0 = (unsigned int)(uintptr_t)lds_dst;   // low 32 = LDS offset
  unsigned long long gb = (unsigned long long)(uintptr_t)gsrc;
  for (int off = threadIdx.x * 16; off < bytes; off += TPB * 16) {
    unsigned int l = lds0 + (unsigned int)off;
    asm volatile("global_load_async_to_lds_b128 %0, %1, %2"
                 :: "v"(l), "v"((unsigned int)off), "s"(gb) : "memory");
  }
  asm volatile("s_wait_asynccnt 0x0" ::: "memory");
}

// bf16 (bits) -> f32 helpers
__device__ __forceinline__ float bf_lo(unsigned int w) {
  union { unsigned int i; float f; } v; v.i = w << 16; return v.f;
}
__device__ __forceinline__ float bf_hi(unsigned int w) {
  union { unsigned int i; float f; } v; v.i = w & 0xffff0000u; return v.f;
}

// ---------------------------------------------------------------------------
// utility: zero a float4 range
__global__ void kZero4(float4* __restrict__ p, int n4) {
  int i = blockIdx.x * blockDim.x + threadIdx.x;
  if (i < n4) p[i] = make_float4(0.f, 0.f, 0.f, 0.f);
}

// ---------------------------------------------------------------------------
// one-time weight pre-conversion to bf16 in the exact LDS image layouts
// W2^T image per layer: out[c*1024 + l*64 + h] = W_lrp[i, h, c, l]
__global__ void kPrepLRP(const float* __restrict__ W, __bf16* __restrict__ out) {
  int t = blockIdx.x * blockDim.x + threadIdx.x;
  if (t >= NLAYERS * 64 * 1024) return;
  int i = t >> 16, t16 = t & 65535;
  int h = t16 & 63, l = (t16 >> 6) & 15, c = t16 >> 10;
  out[t] = (__bf16)W[(size_t)i * 65536 + (h * 64 + c) * 16 + l];
}
// dn1^T image per layer: out[c*128 + j] = dn1_W[i, j, c]
__global__ void kPrepDn1(const float* __restrict__ W, __bf16* __restrict__ out) {
  int t = blockIdx.x * blockDim.x + threadIdx.x;
  if (t >= NLAYERS * 128 * 64) return;
  int i = t >> 13, t13 = t & 8191;
  int j = t13 & 127, c = t13 >> 7;
  out[t] = (__bf16)W[(size_t)i * 8192 + j * 64 + c];
}
// atom_W^T image, K padded 28->32: out[c*32 + kk]
__global__ void kPrepAtom(const float* __restrict__ W, __bf16* __restrict__ out) {
  int t = blockIdx.x * blockDim.x + threadIdx.x;
  if (t >= 64 * 32) return;
  int kk = t & 31, c = t >> 5;
  out[t] = (__bf16)(kk < 28 ? W[kk * HID + c] : 0.0f);
}

// ---------------------------------------------------------------------------
// atom encoder: xbf[n, c] = bf16( node_feat[n,0:28] @ atom_W[28,64] + atom_b )
__global__ __launch_bounds__(TPB)
void kAtom(const float* __restrict__ nf, const __bf16* __restrict__ awbf,
           const float* __restrict__ ab, __bf16* __restrict__ xout) {
  extern __shared__ __attribute__((aligned(32))) char smem_a[];
  __bf16* lw = (__bf16*)smem_a;                 // [64 c][32 k]
  async_stage(lw, awbf, 64 * 32 * 2);
  __syncthreads();

  int warp = threadIdx.x >> 5, lane = threadIdx.x & 31;
  int tile = blockIdx.x * WPB + warp;
  if (tile >= NTILES) return;                   // wave-uniform, EXEC stays full
  int lm = lane & 15, hi = lane >> 4;
  int n0 = tile * 16;

  const float* nr = nf + (size_t)(n0 + lm) * 28;
  v16bf a;
#pragma unroll
  for (int e = 0; e < 16; ++e) {                // A layout: K=e+16*hi
    int K = hi * 16 + e;
    a[e] = (__bf16)(K < 28 ? nr[K] : 0.0f);
  }
  v8f acc[4] = {};
#pragma unroll
  for (int nt = 0; nt < 4; ++nt) {
    v16bf b = *(const v16bf*)(lw + (nt * 16 + lm) * 32 + hi * 16);
    acc[nt] = __builtin_amdgcn_wmma_f32_16x16x32_bf16(
        false, a, false, b, (short)0, acc[nt], false, false);
  }
  float bias[4];
#pragma unroll
  for (int nt = 0; nt < 4; ++nt) bias[nt] = ab[nt * 16 + lm];
#pragma unroll
  for (int r = 0; r < 8; ++r) {                 // C layout: M=r+8*hi, N=lm
    int n = n0 + r + hi * 8;
    __bf16* dst = xout + (size_t)n * HID + lm;
#pragma unroll
    for (int nt = 0; nt < 4; ++nt) dst[nt * 16] = (__bf16)(acc[nt][r] + bias[nt]);
  }
}

// ---------------------------------------------------------------------------
// LRP layer core: h = relu( gathered_T[NPERM,1024] @ W2[1024,64] + b ),
// fused with pooled scatter: pooled[pool_row[p]] += pool_val[p]*h[p].
// x is stored bf16 -> halves the dominant L2 gather traffic.
__global__ __launch_bounds__(TPB)
void kLRP(const unsigned short* __restrict__ xin,   // bf16 bits [N][64]
          const int* __restrict__ n2p_col, const float* __restrict__ n2p_val,
          const float* __restrict__ e2p_val,
          const float* __restrict__ eW, const float* __restrict__ eb,
          const __bf16* __restrict__ wbf, const float* __restrict__ bl,
          const int* __restrict__ pool_row, const float* __restrict__ pool_val,
          float* __restrict__ pooled) {
  extern __shared__ __attribute__((aligned(32))) char smem_l[];
  __bf16* lw = (__bf16*)smem_l;                 // [64 c][1024 k] = 128 KB
  async_stage(lw, wbf, 64 * 1024 * 2);          // async DMA, no VGPR round-trip
  __syncthreads();

  int warp = threadIdx.x >> 5, lane = threadIdx.x & 31;
  int tile = blockIdx.x * WPB + warp;
  if (tile >= NTILES) return;
  int lm = lane & 15, hi = lane >> 4;

  // per-lane erow slice (constant edge-feature row), both 32-wide K halves
  float er[2][16];
#pragma unroll
  for (int hb = 0; hb < 2; ++hb)
#pragma unroll
    for (int e = 0; e < 16; ++e) {
      int h = hb * 32 + hi * 8 + (e < 8 ? e : e + 8);
      er[hb][e] = eW[h] + eb[h];
    }

  int p0 = tile * 16;
  int rowbase = (p0 + lm) * LRP_L;
  v8f acc[4] = {};
  for (int lq = 0; lq < 4; ++lq) {
    int4   c4 = *(const int4*)  (n2p_col + rowbase + lq * 4);
    float4 n4 = *(const float4*)(n2p_val + rowbase + lq * 4);
    float4 e4 = *(const float4*)(e2p_val + rowbase + lq * 4);
    int   cols[4] = {c4.x, c4.y, c4.z, c4.w};
    float nvs[4]  = {n4.x, n4.y, n4.z, n4.w};
    float evs[4]  = {e4.x, e4.y, e4.z, e4.w};
#pragma unroll
    for (int lo = 0; lo < 4; ++lo)              // hide L2 gather latency
      __builtin_prefetch((const char*)xin + (size_t)cols[lo] * 128 + hi * 16, 0, 0);
#pragma unroll
    for (int lo = 0; lo < 4; ++lo) {
      int l = lq * 4 + lo;
      float nv = nvs[lo], ev = evs[lo];
      // whole 128B bf16 row slice for this lane: 4 x 16B chunks
      const uint4* xq =
          (const uint4*)((const char*)xin + (size_t)cols[lo] * 128 + hi * 16);
      uint4 ch[4] = {xq[0], xq[2], xq[4], xq[6]};   // byte offs 0,32,64,96
#pragma unroll
      for (int hb = 0; hb < 2; ++hb) {
        unsigned int w[8] = {ch[hb * 2].x,     ch[hb * 2].y,
                             ch[hb * 2].z,     ch[hb * 2].w,
                             ch[hb * 2 + 1].x, ch[hb * 2 + 1].y,
                             ch[hb * 2 + 1].z, ch[hb * 2 + 1].w};
        v16bf a;
#pragma unroll
        for (int k = 0; k < 8; ++k) {
          a[2 * k]     = (__bf16)(nv * bf_lo(w[k]) + ev * er[hb][2 * k]);
          a[2 * k + 1] = (__bf16)(nv * bf_hi(w[k]) + ev * er[hb][2 * k + 1]);
        }
        int kb = (l * 2 + hb) * 32 + hi * 8;    // B layout: K = 8*hi + (e<8?e:e+8)
#pragma unroll
        for (int nt = 0; nt < 4; ++nt) {
          const __bf16* bp = lw + (nt * 16 + lm) * 1024 + kb;
          v16bf b;
#pragma unroll
          for (int e = 0; e < 16; ++e) b[e] = bp[e < 8 ? e : e + 8];
          acc[nt] = __builtin_amdgcn_wmma_f32_16x16x32_bf16(
              false, a, false, b, (short)0, acc[nt], false, false);
        }
      }
    }
  }

  float bias[4];
#pragma unroll
  for (int nt = 0; nt < 4; ++nt) bias[nt] = bl[nt * 16 + lm];
#pragma unroll
  for (int r = 0; r < 8; ++r) {
    int p = p0 + r + hi * 8;
    int prow = pool_row[p];
    float pv = pool_val[p];
    float* dst = pooled + (size_t)prow * HID + lm;
#pragma unroll
    for (int nt = 0; nt < 4; ++nt) {
      float hv = acc[nt][r] + bias[nt];
      hv = hv > 0.0f ? hv : 0.0f;
      atomicAdd(dst + nt * 16, pv * hv);        // L2-resident atomic scatter
    }
  }
}

// ---------------------------------------------------------------------------
// xbf[n,:] = bf16( pooled[n,:] * ( relu(deg*w0+b0)[128] @ dn1_W[128,64] + b1 ) )
__global__ __launch_bounds__(TPB)
void kFactor(const float* __restrict__ pooled, const float* __restrict__ degs,
             const float* __restrict__ w0, const float* __restrict__ b0,
             const __bf16* __restrict__ w1bf, const float* __restrict__ b1,
             __bf16* __restrict__ xb) {
  extern __shared__ __attribute__((aligned(32))) char smem_f[];
  __bf16* lw = (__bf16*)smem_f;                 // [64 c][128 j] = 16 KB
  float* w0f = (float*)(smem_f + 64 * 128 * 2);
  float* b0f = w0f + 128;
  async_stage(lw, w1bf, 64 * 128 * 2);
  if (threadIdx.x < 128) {
    w0f[threadIdx.x] = w0[threadIdx.x];
    b0f[threadIdx.x] = b0[threadIdx.x];
  }
  __syncthreads();

  int warp = threadIdx.x >> 5, lane = threadIdx.x & 31;
  int tile = blockIdx.x * WPB + warp;
  if (tile >= NTILES) return;
  int lm = lane & 15, hi = lane >> 4;
  int n0 = tile * 16;
  float deg = degs[n0 + lm];

  v8f acc[4] = {};
#pragma unroll
  for (int kc = 0; kc < 4; ++kc) {
    v16bf a;
#pragma unroll
    for (int e = 0; e < 16; ++e) {
      int j = kc * 32 + hi * 16 + e;
      float hv = deg * w0f[j] + b0f[j];
      a[e] = (__bf16)(hv > 0.0f ? hv : 0.0f);
    }
    int kb = kc * 32 + hi * 16;
#pragma unroll
    for (int nt = 0; nt < 4; ++nt) {
      v16bf b = *(const v16bf*)(lw + (nt * 16 + lm) * 128 + kb);
      acc[nt] = __builtin_amdgcn_wmma_f32_16x16x32_bf16(
          false, a, false, b, (short)0, acc[nt], false, false);
    }
  }

  float bias[4];
#pragma unroll
  for (int nt = 0; nt < 4; ++nt) bias[nt] = b1[nt * 16 + lm];
#pragma unroll
  for (int r = 0; r < 8; ++r) {
    int n = n0 + r + hi * 8;
    const float* pr = pooled + (size_t)n * HID + lm;
    __bf16* dst = xb + (size_t)n * HID + lm;
#pragma unroll
    for (int nt = 0; nt < 4; ++nt) {
      float f = acc[nt][r] + bias[nt];
      dst[nt * 16] = (__bf16)(pr[nt * 16] * f);
    }
  }
}

// ---------------------------------------------------------------------------
// global mean pool (x stored as bf16 bits)
__global__ void kPool(const unsigned short* __restrict__ x,
                      const int* __restrict__ batch,
                      float* __restrict__ out, float* __restrict__ cnt) {
  int idx = blockIdx.x * blockDim.x + threadIdx.x;
  if (idx >= N_NODES * HID) return;
  int n = idx >> 6, c = idx & 63;
  int g = batch[n];
  atomicAdd(&out[g * HID + c], bf_lo((unsigned int)x[idx]));
  if (c == 0) atomicAdd(&cnt[g], 1.0f);
}

__global__ void kDiv(float* __restrict__ out, const float* __restrict__ cnt) {
  int idx = blockIdx.x * blockDim.x + threadIdx.x;
  if (idx >= NGRAPH * HID) return;
  out[idx] /= fmaxf(cnt[idx >> 6], 1.0f);
}

// ---------------------------------------------------------------------------
extern "C" void kernel_launch(void* const* d_in, const int* in_sizes, int n_in,
                              void* d_out, int out_size, void* d_ws, size_t ws_size,
                              hipStream_t stream) {
  (void)in_sizes; (void)n_in; (void)out_size; (void)ws_size;
  const float* node_feat = (const float*)d_in[0];
  const float* degs      = (const float*)d_in[1];
  const int*   batch     = (const int*)  d_in[2];
  const int*   n2p_col   = (const int*)  d_in[4];
  const float* n2p_val   = (const float*)d_in[5];
  const float* e2p_val   = (const float*)d_in[8];   // e2p_col dead: rows identical
  const int*   pool_row  = (const int*)  d_in[9];
  const float* pool_val  = (const float*)d_in[11];
  const float* atom_W    = (const float*)d_in[12];
  const float* atom_b    = (const float*)d_in[13];
  const float* edge_W    = (const float*)d_in[14];
  const float* edge_b    = (const float*)d_in[15];
  const float* W_lrp     = (const float*)d_in[16];
  const float* b_lrp     = (const float*)d_in[17];
  const float* dn0_W     = (const float*)d_in[18];
  const float* dn0_b     = (const float*)d_in[19];
  const float* dn1_W     = (const float*)d_in[20];
  const float* dn1_b     = (const float*)d_in[21];

  const size_t NH = (size_t)N_NODES * HID;
  float*  pooled  = (float*)d_ws;                   // f32 (atomic target)
  __bf16* xbfA    = (__bf16*)(pooled + NH);         // bf16 activation ping
  __bf16* xbfB    = xbfA + NH;                      // bf16 activation pong
  float*  cnts    = (float*)(xbfB + NH);
  __bf16* wlrp_bf = (__bf16*)(cnts + NGRAPH);       // 4 x 128KB LDS images
  __bf16* dn1_bf  = wlrp_bf + NLAYERS * 64 * 1024;  // 4 x 16KB images
  __bf16* atom_bf = dn1_bf + NLAYERS * 128 * 64;    // 4KB image

  hipFuncSetAttribute((const void*)kLRP,
                      hipFuncAttributeMaxDynamicSharedMemorySize, 131072);

  // one-time bf16 weight images (exact LDS layouts)
  kPrepLRP <<<(NLAYERS * 65536 + TPB - 1) / TPB, TPB, 0, stream>>>(W_lrp, wlrp_bf);
  kPrepDn1 <<<(NLAYERS * 8192  + TPB - 1) / TPB, TPB, 0, stream>>>(dn1_W, dn1_bf);
  kPrepAtom<<<(2048            + TPB - 1) / TPB, TPB, 0, stream>>>(atom_W, atom_bf);

  int gB = (NTILES + WPB - 1) / WPB;  // 782 blocks of 8 wave32-tiles

  kAtom<<<gB, TPB, 64 * 32 * 2, stream>>>(node_feat, atom_bf, atom_b, xbfA);

  __bf16* cur = xbfA;
  __bf16* oth = xbfB;
  for (int i = 0; i < NLAYERS; ++i) {
    int nz4 = N_NODES * HID / 4;
    kZero4<<<(nz4 + TPB - 1) / TPB, TPB, 0, stream>>>((float4*)pooled, nz4);
    kLRP<<<gB, TPB, 131072, stream>>>(
        (const unsigned short*)cur, n2p_col, n2p_val, e2p_val, edge_W, edge_b,
        wlrp_bf + (size_t)i * 64 * 1024, b_lrp + i * HID,
        pool_row, pool_val, pooled);
    kFactor<<<gB, TPB, 64 * 128 * 2 + 256 * 4, stream>>>(
        pooled, degs, dn0_W + i * 2 * HID, dn0_b + i * 2 * HID,
        dn1_bf + (size_t)i * 128 * 64, dn1_b + i * HID, oth);
    __bf16* t = cur; cur = oth; oth = t;
  }

  int no4 = NGRAPH * HID / 4;
  kZero4<<<(no4 + TPB - 1) / TPB, TPB, 0, stream>>>((float4*)d_out, no4);
  kZero4<<<1, TPB, 0, stream>>>((float4*)cnts, NGRAPH / 4);
  int np = N_NODES * HID;
  kPool<<<(np + TPB - 1) / TPB, TPB, 0, stream>>>(
      (const unsigned short*)cur, batch, (float*)d_out, cnts);
  kDiv<<<(NGRAPH * HID + TPB - 1) / TPB, TPB, 0, stream>>>((float*)d_out, cnts);
}
